// MultiHeadAttentionQuantum_65481071396986
// MI455X (gfx1250) — matrix operand order repaired and checked by reading, not compiled, not run on hardware
//
#include <hip/hip_runtime.h>
#include <hip/hip_bf16.h>

typedef __attribute__((ext_vector_type(16))) _Float16 v16h;
typedef __attribute__((ext_vector_type(8)))  _Float16 v8h;
typedef __attribute__((ext_vector_type(8)))  float    v8f;

#define QSTRIDE 66   // padded row stride (halves) for staged Q tiles (bank-conflict avoidance)
#define WSTRIDE 72   // padded row stride (halves) for staged W (144B rows, 16B aligned chunks)

union V16H { v16h v; _Float16 h[16]; };

// hardware cosine: V_COS_F32 computes cos(2*pi*x) -> prescale by 1/(2*pi)
__device__ __forceinline__ float fcos(float x) {
    return __builtin_amdgcn_cosf(x * 0.15915494309189535f);
}

__global__ __launch_bounds__(256)
void qattn_fused_kernel(const float* __restrict__ x,
                        const float* __restrict__ W,
                        const float* __restrict__ bvec,
                        const float* __restrict__ qp,
                        float* __restrict__ out) {
    __shared__ _Float16 sW[64 * WSTRIDE];            // 9216 B
    __shared__ _Float16 sQ[8][16 * QSTRIDE];         // 16896 B

    const int tid  = threadIdx.x;
    const int wave = tid >> 5;        // 8 waves / block (wave32)
    const int lane = tid & 31;
    const int hi   = lane >> 4;       // half-wave select
    const int lan  = lane & 15;

    // ---- stage W_combine (64x64 f32 -> f16, padded rows), cooperative ----
    for (int i = tid; i < 64 * 64; i += 256) {
        int r = i >> 6, c = i & 63;
        sW[r * WSTRIDE + c] = (_Float16)W[i];
    }

    const float p0 = qp[0], p1 = qp[1], p3 = qp[3], p4 = qp[4];
    const float cp1 = fcos(p1), cp4 = fcos(p4);

    const int rowTile = blockIdx.x * 128 + wave * 16;   // 16 rows per wave

    // ---- closed-form "quantum circuit": cosines into LDS as f16 ----
    // 128 groups-of-8 per wave tile; 4 per lane, coalesced 32B x-loads
    #pragma unroll
    for (int gi = 0; gi < 4; ++gi) {
        int glin = lane + 32 * gi;     // 0..127
        int row  = glin >> 3;          // 0..15
        int grp  = glin & 7;           // head group within row
        const float4* xp4 = (const float4*)(x + (size_t)(rowTile + row) * 64 + grp * 8);
        float4 lo = xp4[0];
        float4 hv = xp4[1];
        float c1 = fcos(lo.y), c4 = fcos(hv.x);
        _Float16* dst = &sQ[wave][row * QSTRIDE + grp * 8];
        dst[0] = (_Float16)(c4 * fcos(lo.x + p0));   // <Z0> = cos(x4)cos(x0+p0)
        dst[1] = (_Float16)(c1);                     // <Z1> = cos(x1)
        dst[2] = (_Float16)(c1 * fcos(lo.z));        // <Z2> = cos(x1)cos(x2)
        dst[3] = (_Float16)(cp1 * fcos(lo.w));       // <Z3> = cos(p1)cos(x3)
        dst[4] = (_Float16)(c4);                     // <Z4> = cos(x4)
        dst[5] = (_Float16)(fcos(hv.y));             // <Z5> = cos(x5)
        dst[6] = (_Float16)(fcos(hv.z + p3));        // <Z6> = cos(x6+p3)
        dst[7] = (_Float16)(cp4 * fcos(hv.w));       // <Z7> = cos(p4)cos(x7)
    }

    __syncthreads();

    // ---- accumulators seeded with bias (column = jt*16 + lan, shared by all 8 rows) ----
    v8f acc[4];
    #pragma unroll
    for (int jt = 0; jt < 4; ++jt) {
        float bb = bvec[jt * 16 + lan];
        #pragma unroll
        for (int r = 0; r < 8; ++r) acc[jt][r] = bb;
    }

    // ---- WMMA GEMM: D[16,64] = Q[16,64] * W^T + bias ----
    #pragma unroll
    for (int kc = 0; kc < 2; ++kc) {
        // A fragment (16-bit A 16x32 layout): lane holds row M=lan,
        // half i -> K = 32*kc + 8*hi + (i&7) + 16*(i>>3)
        V16H A;
        const _Float16* qrow = &sQ[wave][lan * QSTRIDE + 32 * kc + 8 * hi];
        #pragma unroll
        for (int p = 0; p < 8; ++p) {
            int k = 2 * (p & 3) + 16 * (p >> 2);
            A.h[2 * p]     = qrow[k];
            A.h[2 * p + 1] = qrow[k + 1];
        }
        #pragma unroll
        for (int jt = 0; jt < 4; ++jt) {
            // B fragment (16-bit B 32x16): lane holds N=lan, K = kc*32 + 8*hi + (i&7) + 16*(i>>3)
            const _Float16* wrow = &sW[(jt * 16 + lan) * WSTRIDE + kc * 32 + 8 * hi];
            V16H B;
            *(v8h*)&B.h[0] = *(const v8h*)(wrow);
            *(v8h*)&B.h[8] = *(const v8h*)(wrow + 16);
            acc[jt] = __builtin_amdgcn_wmma_f32_16x16x32_f16(
                false, A.v, false, B.v, (short)0, acc[jt], false, false);
        }
    }

    // ---- epilogue: C/D layout M = r + 8*hi, N = lan; coalesced f32 stores ----
    #pragma unroll
    for (int jt = 0; jt < 4; ++jt) {
        int col = jt * 16 + lan;
        #pragma unroll
        for (int r = 0; r < 8; ++r) {
            out[(size_t)(rowTile + r + 8 * hi) * 64 + col] = acc[jt][r];
        }
    }
}

extern "C" void kernel_launch(void* const* d_in, const int* in_sizes, int n_in,
                              void* d_out, int out_size, void* d_ws, size_t ws_size,
                              hipStream_t stream) {
    const float* x  = (const float*)d_in[0];   // (8,1024,64) f32
    const float* W  = (const float*)d_in[1];   // (64,64) f32
    const float* b  = (const float*)d_in[2];   // (64,) f32
    const float* qp = (const float*)d_in[3];   // (6,) f32
    float* out = (float*)d_out;                // (8,1024,64) f32

    dim3 grid(64);     // 64 blocks * 8 waves * 16 rows = 8192 rows exactly
    dim3 block(256);
    qattn_fused_kernel<<<grid, block, 0, stream>>>(x, W, b, qp, out);
}
